// Q_Encoder_17609365913770
// MI455X (gfx1250) — compile-verified, matrix operand order checked by reading, not run on hardware
//
#include <hip/hip_runtime.h>
#include <hip/hip_bf16.h>
#include <cstdint>
#include <cstddef>

// Problem dims (compile-time constants from the reference)
#define BDIM 1024
#define TDIM 40
#define EDIM 512
#define HDIM 1024
#define LDIM 256
#define ADIM 2048
#define KC   (EDIM + HDIM)   // 1536 concatenated K for the recurrent GEMM
#define G4H  (4 * HDIM)      // 4096

typedef __attribute__((ext_vector_type(16))) __bf16 v16bf;
typedef __attribute__((ext_vector_type(8)))  float  v8f;
typedef unsigned int u32x4 __attribute__((ext_vector_type(4)));

// ---------------------------------------------------------------------------
// Fragment load for CDNA5 16-bit WMMA A/B operand from a row-major matrix.
// Lane l holds row (l&15); K elements {kh..kh+7} then {kh+16..kh+23}, kh=8*(l>>4).
// Two 16-byte contiguous loads per lane.
// ---------------------------------------------------------------------------
__device__ __forceinline__ v16bf ldfrag(const __bf16* p) {
    union { v16bf v; u32x4 q[2]; } u;
    u.q[0] = *reinterpret_cast<const u32x4*>(p);
    u.q[1] = *reinterpret_cast<const u32x4*>(p + 16);
    return u.v;
}

// ---------------------------------------------------------------------------
// Generic GEMM: D[M,N] = A[M,K] * B[N,K]^T + bias[N]   (A,B bf16 row-major,
// D f32 row-major). Block = 256 threads = 8 waves arranged 2(M) x 4(N),
// wave tile 64x64 (4x4 WMMA 16x16x32 tiles -> 16 WMMA per 8 fragment loads),
// block tile 128x256. All dims divisible -> no guards, EXEC all-ones.
// ---------------------------------------------------------------------------
__global__ __launch_bounds__(256, 1)
void gemm_bf16_kernel(const __bf16* __restrict__ Am, const __bf16* __restrict__ Bm,
                      const float* __restrict__ bias, float* __restrict__ D,
                      int K, int N)
{
    const int lane = threadIdx.x & 31;
    const int wave = threadIdx.x >> 5;
    const int wm   = wave & 1;          // 2 waves along M
    const int wn   = wave >> 1;         // 4 waves along N
    const int r    = lane & 15;
    const int kh   = (lane >> 4) << 3;  // 0 or 8

    const int m_base = blockIdx.y * 128 + wm * 64;
    const int n_base = blockIdx.x * 256 + wn * 64;

    v8f acc[4][4];
#pragma unroll
    for (int nj = 0; nj < 4; ++nj) {
        const float bv = bias ? bias[n_base + 16 * nj + r] : 0.0f;
        v8f z = {bv, bv, bv, bv, bv, bv, bv, bv};
#pragma unroll
        for (int mi = 0; mi < 4; ++mi) acc[mi][nj] = z;
    }

    const __bf16* ap = Am + (size_t)(m_base + r) * K + kh;
    const __bf16* bp = Bm + (size_t)(n_base + r) * K + kh;

#pragma unroll 2
    for (int k = 0; k < K; k += 32) {
        v16bf af[4], bf[4];
#pragma unroll
        for (int mi = 0; mi < 4; ++mi)
            af[mi] = ldfrag(ap + (size_t)(16 * mi) * K + k);
#pragma unroll
        for (int nj = 0; nj < 4; ++nj)
            bf[nj] = ldfrag(bp + (size_t)(16 * nj) * K + k);
#pragma unroll
        for (int mi = 0; mi < 4; ++mi)
#pragma unroll
            for (int nj = 0; nj < 4; ++nj)
                acc[mi][nj] = __builtin_amdgcn_wmma_f32_16x16x32_bf16(
                    false, af[mi], false, bf[nj], (short)0, acc[mi][nj], false, false);
    }

    const int row_off = (lane >> 4) * 8;   // C/D layout: vgpr v -> M = v + 8*(lane>=16)
#pragma unroll
    for (int mi = 0; mi < 4; ++mi) {
#pragma unroll
        for (int nj = 0; nj < 4; ++nj) {
            float* dp = D + (size_t)(m_base + 16 * mi + row_off) * N
                          + (n_base + 16 * nj + r);
#pragma unroll
            for (int v = 0; v < 8; ++v)
                dp[(size_t)v * N] = acc[mi][nj][v];
        }
    }
}

// ---------------------------------------------------------------------------
// weight_norm row kernel: out_bf16[row,:] = g[row] * v[row,:] / ||v[row,:]||
// ---------------------------------------------------------------------------
__global__ void rownorm_kernel(const float* __restrict__ v, const float* __restrict__ g,
                               __bf16* __restrict__ out, int cols)
{
    const int row = blockIdx.x;
    const float* vr = v + (size_t)row * cols;
    float s = 0.0f;
    for (int i = threadIdx.x; i < cols; i += 256) { float x = vr[i]; s += x * x; }
    __shared__ float red[256];
    red[threadIdx.x] = s;
    __syncthreads();
    for (int st = 128; st > 0; st >>= 1) {
        if (threadIdx.x < st) red[threadIdx.x] += red[threadIdx.x + st];
        __syncthreads();
    }
    const float scale = g[row] / sqrtf(red[0]);
    __bf16* orow = out + (size_t)row * cols;
    for (int i = threadIdx.x; i < cols; i += 256)
        orow[i] = (__bf16)(vr[i] * scale);
}

// ---------------------------------------------------------------------------
// Build Wcat[4H, E+H] = [W_ih | W_hh] in bf16, and bsum = b_ih + b_hh.
// ---------------------------------------------------------------------------
__global__ void catw_kernel(const float* __restrict__ Wih, const float* __restrict__ Whh,
                            const float* __restrict__ bih, const float* __restrict__ bhh,
                            __bf16* __restrict__ Wcat, float* __restrict__ bsum)
{
    const int row = blockIdx.x;                 // 0..4095
    const float* a = Wih + (size_t)row * EDIM;
    const float* b = Whh + (size_t)row * HDIM;
    __bf16* o = Wcat + (size_t)row * KC;
    for (int i = threadIdx.x; i < EDIM; i += 256) o[i] = (__bf16)a[i];
    for (int i = threadIdx.x; i < HDIM; i += 256) o[EDIM + i] = (__bf16)b[i];
    if (threadIdx.x == 0) bsum[row] = bih[row] + bhh[row];
}

__global__ void cvt_kernel(const float* __restrict__ in, __bf16* __restrict__ out, int n)
{
    const int i = blockIdx.x * 256 + threadIdx.x;
    if (i < n) out[i] = (__bf16)in[i];
}

// Seed the h-half of xcat from the f32 h0 produced by the init GEMM.
__global__ void seed_h_kernel(const float* __restrict__ h0, __bf16* __restrict__ xcat)
{
    const int idx = blockIdx.x * 256 + threadIdx.x;   // < B*H
    const int b = idx >> 10;
    const int j = idx & (HDIM - 1);
    xcat[(size_t)b * KC + EDIM + j] = (__bf16)h0[idx];
}

// ---------------------------------------------------------------------------
// Pack embedding half only: xcat[b, 0:E] = bf16(embed[cap[b,t], :])
// (the h-half is written by cell_kernel of the previous step)
// ---------------------------------------------------------------------------
__global__ void pack_kernel(const int* __restrict__ cap, const float* __restrict__ embed,
                            __bf16* __restrict__ xcat, int t)
{
    const int idx = blockIdx.x * 256 + threadIdx.x;   // < B*E
    const int b = idx >> 9;            // /EDIM
    const int j = idx & (EDIM - 1);
    const int tok = cap[b * TDIM + t];
    xcat[(size_t)b * KC + j] = (__bf16)embed[(size_t)tok * EDIM + j];
}

// ---------------------------------------------------------------------------
// LSTM cell elementwise (gate order i,f,g,o). Writes bf16 h straight into the
// h-half of xcat (next step's A operand) and captures h at t == cap_len-1.
// ---------------------------------------------------------------------------
__global__ void cell_kernel(const float* __restrict__ gates, float* __restrict__ c,
                            __bf16* __restrict__ xcat, const int* __restrict__ cap_len,
                            __bf16* __restrict__ hlast, int t)
{
    const int idx = blockIdx.x * 256 + threadIdx.x;   // < B*H
    const int b = idx >> 10;
    const int j = idx & (HDIM - 1);
    const float* gr = gates + (size_t)b * G4H;
    const float ig = gr[j];
    const float fg = gr[HDIM + j];
    const float gg = gr[2 * HDIM + j];
    const float og = gr[3 * HDIM + j];
    const float si = 1.0f / (1.0f + __expf(-ig));
    const float sf = 1.0f / (1.0f + __expf(-fg));
    const float so = 1.0f / (1.0f + __expf(-og));
    const float cn = sf * c[idx] + si * tanhf(gg);
    const float hn = so * tanhf(cn);
    c[idx] = cn;
    xcat[(size_t)b * KC + EDIM + j] = (__bf16)hn;
    if (cap_len[b] - 1 == t) hlast[idx] = (__bf16)hn;
}

// ---------------------------------------------------------------------------
extern "C" void kernel_launch(void* const* d_in, const int* in_sizes, int n_in,
                              void* d_out, int out_size, void* d_ws, size_t ws_size,
                              hipStream_t stream)
{
    (void)in_sizes; (void)n_in; (void)out_size; (void)ws_size;

    const int*   cap     = (const int*)  d_in[0];
    const int*   cap_len = (const int*)  d_in[1];
    const float* feat    = (const float*)d_in[2];
    const float* embed   = (const float*)d_in[3];
    const float* W_ih    = (const float*)d_in[4];
    const float* W_hh    = (const float*)d_in[5];
    const float* b_ih    = (const float*)d_in[6];
    const float* b_hh    = (const float*)d_in[7];
    const float* v_ih0   = (const float*)d_in[8];
    const float* g_ih0   = (const float*)d_in[9];
    const float* b_ih0   = (const float*)d_in[10];
    const float* v_ic0   = (const float*)d_in[11];
    const float* g_ic0   = (const float*)d_in[12];
    const float* b_ic0   = (const float*)d_in[13];
    const float* v_mu    = (const float*)d_in[14];
    const float* g_mu    = (const float*)d_in[15];
    const float* b_mu    = (const float*)d_in[16];
    const float* v_s2    = (const float*)d_in[17];
    const float* g_s2    = (const float*)d_in[18];
    const float* b_s2    = (const float*)d_in[19];

    float* out = (float*)d_out;   // mu [B,L] then sigma2 [B,L]

    // --- workspace layout (256B aligned) ---
    char* ws = (char*)d_ws;
    size_t off = 0;
    auto take = [&](size_t bytes) -> char* {
        char* p = ws + off;
        off = (off + bytes + 255) & ~(size_t)255;
        return p;
    };
    __bf16* Wcat    = (__bf16*)take((size_t)G4H * KC * 2);
    float*  bsum    = (float*) take((size_t)G4H * 4);
    __bf16* Wn_ih0  = (__bf16*)take((size_t)HDIM * ADIM * 2);
    __bf16* Wn_ic0  = (__bf16*)take((size_t)HDIM * ADIM * 2);
    __bf16* Wn_mu   = (__bf16*)take((size_t)LDIM * HDIM * 2);
    __bf16* Wn_s2   = (__bf16*)take((size_t)LDIM * HDIM * 2);
    __bf16* featbf  = (__bf16*)take((size_t)BDIM * ADIM * 2);
    float*  h0f     = (float*) take((size_t)BDIM * HDIM * 4);
    float*  c       = (float*) take((size_t)BDIM * HDIM * 4);
    __bf16* xcat    = (__bf16*)take((size_t)BDIM * KC * 2);
    float*  gates   = (float*) take((size_t)BDIM * G4H * 4);
    __bf16* hlast   = (__bf16*)take((size_t)BDIM * HDIM * 2);

    // --- 1. weight prep (re-done every call: deterministic, no caching) ---
    rownorm_kernel<<<HDIM, 256, 0, stream>>>(v_ih0, g_ih0, Wn_ih0, ADIM);
    rownorm_kernel<<<HDIM, 256, 0, stream>>>(v_ic0, g_ic0, Wn_ic0, ADIM);
    rownorm_kernel<<<LDIM, 256, 0, stream>>>(v_mu,  g_mu,  Wn_mu,  HDIM);
    rownorm_kernel<<<LDIM, 256, 0, stream>>>(v_s2,  g_s2,  Wn_s2,  HDIM);
    catw_kernel<<<G4H, 256, 0, stream>>>(W_ih, W_hh, b_ih, b_hh, Wcat, bsum);
    cvt_kernel<<<(BDIM * ADIM) / 256, 256, 0, stream>>>(feat, featbf, BDIM * ADIM);

    // --- 2. initial state: h0/c0 = feat @ Wn^T + b ---
    gemm_bf16_kernel<<<dim3(HDIM / 256, BDIM / 128), 256, 0, stream>>>(
        featbf, Wn_ih0, b_ih0, h0f, ADIM, HDIM);
    gemm_bf16_kernel<<<dim3(HDIM / 256, BDIM / 128), 256, 0, stream>>>(
        featbf, Wn_ic0, b_ic0, c, ADIM, HDIM);
    seed_h_kernel<<<(BDIM * HDIM) / 256, 256, 0, stream>>>(h0f, xcat);

    // --- 3. recurrent loop ---
    for (int t = 0; t < TDIM; ++t) {
        pack_kernel<<<(BDIM * EDIM) / 256, 256, 0, stream>>>(cap, embed, xcat, t);
        gemm_bf16_kernel<<<dim3(G4H / 256, BDIM / 128), 256, 0, stream>>>(
            xcat, Wcat, bsum, gates, KC, G4H);
        cell_kernel<<<(BDIM * HDIM) / 256, 256, 0, stream>>>(
            gates, c, xcat, cap_len, hlast, t);
    }

    // --- 4. heads: mu and sigma2 straight into d_out ---
    gemm_bf16_kernel<<<dim3(LDIM / 256, BDIM / 128), 256, 0, stream>>>(
        hlast, Wn_mu, b_mu, out, HDIM, LDIM);
    gemm_bf16_kernel<<<dim3(LDIM / 256, BDIM / 128), 256, 0, stream>>>(
        hlast, Wn_s2, b_s2, out + (size_t)BDIM * LDIM, HDIM, LDIM);
}
